// LSTMModel_79620103733595
// MI455X (gfx1250) — matrix operand order repaired and checked by reading, not compiled, not run on hardware
//
#include <hip/hip_runtime.h>

// Fully-fused persistent 2-layer LSTM + FC for MI455X (gfx1250, wave32).
// One workgroup owns 32 batch rows for the entire T-step recurrence.
// All weights + h/c state live in LDS; HBM traffic = x (4MB) + out (8KB).
// Recurrent GEMMs run on v_wmma_f32_16x16x32_f16 (f16 in, f32 accumulate).

#define H       50
#define NPAD    208    // 4H=200 gate columns padded to 13 N-tiles of 16
#define KH      64     // padded K per hidden vector
#define HS      72     // h-buffer row stride in halves (bank spread)
#define W0S     72     // layer0 weight row stride (halves)
#define W1S     136    // layer1 weight row stride (halves), K=128 used
#define CS      52     // c-buffer row stride (floats)
#define BT      32     // batch rows per workgroup
#define NTHREADS 256
#define NWAVES  8

// LDS byte offsets (all 16B aligned)
#define OFF_W1   0              // 208*136*2 = 56576
#define OFF_W0   56576          // 208*72*2  = 29952
#define OFF_H0A  86528          // 32*72*2   = 4608
#define OFF_H0B  91136
#define OFF_H1A  95744
#define OFF_H1B  100352
#define OFF_C0   104960         // 32*52*4   = 6656
#define OFF_C1   111616
#define OFF_H1F  118272
#define OFF_B0   124928         // 208*4
#define OFF_B1   125760
#define OFF_WFC  126592         // 64*4
#define SMEM_BYTES 126848

typedef __attribute__((ext_vector_type(16))) _Float16 v16h;
typedef __attribute__((ext_vector_type(8)))  float    v8f;

union Frag16 { uint4 u[2]; v16h v; };

// A fragment (16x32 f16). Per ISA: lanes 0-15 -> M=lane, K = {0..7, 16..23};
// lanes 16-31 -> M=lane-16, K = {8..15, 24..31}. Two ds_load_b128 per lane.
__device__ __forceinline__ v16h load_a(const _Float16* row0, int strideH, int k0, int lane) {
  int m  = lane & 15;
  int hi = lane >> 4;
  const _Float16* p = row0 + m * strideH + k0 + hi * 8;
  Frag16 f;
  f.u[0] = *(const uint4*)p;
  f.u[1] = *(const uint4*)(p + 16);
  return f.v;
}

// B fragment (32x16 f16), source stored as B^T rows (n-major, k contiguous).
// Lanes 0-15 -> N=lane, K = {0..7, 16..23}; lanes 16-31 -> N=lane-16, K = {8..15, 24..31}.
__device__ __forceinline__ v16h load_b(const _Float16* row0, int strideH, int k0, int lane) {
  int n  = lane & 15;
  int hi = lane >> 4;
  const _Float16* p = row0 + n * strideH + k0 + hi * 8;
  Frag16 f;
  f.u[0] = *(const uint4*)p;
  f.u[1] = *(const uint4*)(p + 16);
  return f.v;
}

__device__ __forceinline__ float fsig(float x) {
  float e = __builtin_amdgcn_exp2f(-1.4426950408889634f * x);
  return __builtin_amdgcn_rcpf(1.0f + e);
}
__device__ __forceinline__ float ftanh(float x) {
  float e = __builtin_amdgcn_exp2f(-2.8853900817779268f * x);
  return 2.0f * __builtin_amdgcn_rcpf(1.0f + e) - 1.0f;
}

// Gate columns are interleaved n = 4*j + q (q: 0=i,1=f,2=g,3=o), so each
// 4-lane group holds all four gates of unit j for 8 batch rows -> full LSTM
// cell update via 4 lane shuffles, no cross-wave traffic.
__device__ __forceinline__ void lstm_update(v8f acc, float* cbuf, _Float16* hout,
                                            float* hf32, int mBase, int nBase, int lane) {
  int l15 = lane & 15;
  int q = l15 & 3;
  int j = (nBase + l15) >> 2;
  int srcBase = lane & ~3;
  int mhalf = lane >> 4;
#pragma unroll
  for (int r = 0; r < 8; ++r) {
    float v  = acc[r];
    float iv = __shfl(v, srcBase + 0, 32);
    float fv = __shfl(v, srcBase + 1, 32);
    float gv = __shfl(v, srcBase + 2, 32);
    float ov = __shfl(v, srcBase + 3, 32);
    if (q == 0 && j < H) {
      iv = fsig(iv); fv = fsig(fv); ov = fsig(ov); gv = ftanh(gv);
      int b = mBase + r + 8 * mhalf;
      float c = cbuf[b * CS + j];
      c = fv * c + iv * gv;
      cbuf[b * CS + j] = c;
      float h = ov * ftanh(c);
      hout[b * HS + j] = (_Float16)h;
      if (hf32) hf32[b * CS + j] = h;
    }
  }
}

__global__ __launch_bounds__(NTHREADS)
void lstm2_fused_kernel(const float* __restrict__ x,
                        const float* __restrict__ Wih0, const float* __restrict__ Whh0,
                        const float* __restrict__ bih0, const float* __restrict__ bhh0,
                        const float* __restrict__ Wih1, const float* __restrict__ Whh1,
                        const float* __restrict__ bih1, const float* __restrict__ bhh1,
                        const float* __restrict__ Wfc,  const float* __restrict__ bfc,
                        float* __restrict__ out, int T) {
  extern __shared__ char smem[];
  _Float16* w1  = (_Float16*)(smem + OFF_W1);
  _Float16* w0  = (_Float16*)(smem + OFF_W0);
  _Float16* h0a = (_Float16*)(smem + OFF_H0A);
  _Float16* h0b = (_Float16*)(smem + OFF_H0B);
  _Float16* h1a = (_Float16*)(smem + OFF_H1A);
  _Float16* h1b = (_Float16*)(smem + OFF_H1B);
  float* c0    = (float*)(smem + OFF_C0);
  float* c1    = (float*)(smem + OFF_C1);
  float* h1f   = (float*)(smem + OFF_H1F);
  float* bias0 = (float*)(smem + OFF_B0);
  float* bias1 = (float*)(smem + OFF_B1);
  float* wfc   = (float*)(smem + OFF_WFC);

  const int tid   = threadIdx.x;
  const int lane  = tid & 31;
  const int wave  = tid >> 5;
  const int bBase = blockIdx.x * BT;

  // ---- stage weights f32->f16 in gate-interleaved layout ----
  // Layer0 B^T rows: k<50 -> W_hh0[g][k]; k==50 -> W_ih0[g] (x rides in pad slot)
  for (int idx = tid; idx < NPAD * KH; idx += NTHREADS) {
    int n = idx >> 6, k = idx & 63;
    int j = n >> 2, q = n & 3;
    float v = 0.0f;
    if (j < H) {
      int g = q * H + j;
      if (k < H) v = Whh0[g * H + k];
      else if (k == H) v = Wih0[g];
    }
    w0[n * W0S + k] = (_Float16)v;
  }
  // Layer1 B^T rows: K=128 concat [W_ih1 | pad | W_hh1 | pad]
  for (int idx = tid; idx < NPAD * 128; idx += NTHREADS) {
    int n = idx >> 7, k = idx & 127;
    int j = n >> 2, q = n & 3;
    float v = 0.0f;
    if (j < H) {
      int g = q * H + j;
      if (k < H) v = Wih1[g * H + k];
      else if (k >= KH && k < KH + H) v = Whh1[g * H + (k - KH)];
    }
    w1[n * W1S + k] = (_Float16)v;
  }
  for (int n = tid; n < NPAD; n += NTHREADS) {
    int j = n >> 2, q = n & 3;
    float v0 = 0.0f, v1 = 0.0f;
    if (j < H) { int g = q * H + j; v0 = bih0[g] + bhh0[g]; v1 = bih1[g] + bhh1[g]; }
    bias0[n] = v0; bias1[n] = v1;
  }
  for (int idx = tid; idx < BT * HS; idx += NTHREADS) {
    h0a[idx] = (_Float16)0.0f; h0b[idx] = (_Float16)0.0f;
    h1a[idx] = (_Float16)0.0f; h1b[idx] = (_Float16)0.0f;
  }
  for (int idx = tid; idx < BT * CS; idx += NTHREADS) {
    c0[idx] = 0.0f; c1[idx] = 0.0f; h1f[idx] = 0.0f;
  }
  if (tid < KH) wfc[tid] = (tid < H) ? Wfc[tid] : 0.0f;
  __syncthreads();

  const int NTILES = 2 * (NPAD / 16);   // 2 M-tiles x 13 N-tiles = 26

  for (int t = 0; t < T; ++t) {
    _Float16* A0 = (t & 1) ? h0b : h0a;   // h0 prev (A operand, col 50 = x_t)
    _Float16* D0 = (t & 1) ? h0a : h0b;   // h0 new
    _Float16* A1 = (t & 1) ? h1b : h1a;   // h1 prev
    _Float16* D1 = (t & 1) ? h1a : h1b;   // h1 new

    if (tid < BT) {
      A0[tid * HS + H] = (_Float16)x[(size_t)(bBase + tid) * T + t];
      if (t + 1 < T) __builtin_prefetch(&x[(size_t)(bBase + tid) * T + t + 1], 0, 1);
    }
    __syncthreads();

    // ---- layer 0: gates = [h0_prev | x] @ W0cat, bias in C ----
    for (int tile = wave; tile < NTILES; tile += NWAVES) {
      int mBase = (tile & 1) * 16;
      int nBase = (tile >> 1) * 16;
      float bb = bias0[nBase + (lane & 15)];
      v8f acc = {bb, bb, bb, bb, bb, bb, bb, bb};
#pragma unroll
      for (int kc = 0; kc < 2; ++kc) {
        v16h a = load_a(A0 + mBase * HS, HS, kc * 32, lane);
        v16h b = load_b(w0 + nBase * W0S, W0S, kc * 32, lane);
        acc = __builtin_amdgcn_wmma_f32_16x16x32_f16(false, a, false, b,
                                                     (short)0, acc, false, false);
      }
      lstm_update(acc, c0, D0, nullptr, mBase, nBase, lane);
    }
    __syncthreads();

    // ---- layer 1: gates = [h0_cur | h1_prev] @ W1cat (K=128), bias in C ----
    float* hf = (t == T - 1) ? h1f : nullptr;
    for (int tile = wave; tile < NTILES; tile += NWAVES) {
      int mBase = (tile & 1) * 16;
      int nBase = (tile >> 1) * 16;
      float bb = bias1[nBase + (lane & 15)];
      v8f acc = {bb, bb, bb, bb, bb, bb, bb, bb};
#pragma unroll
      for (int kc = 0; kc < 4; ++kc) {
        const _Float16* src = (kc < 2) ? D0 : A1;
        int kk = (kc & 1) * 32;
        v16h a = load_a(src + mBase * HS, HS, kk, lane);
        v16h b = load_b(w1 + nBase * W1S, W1S, kc * 32, lane);
        acc = __builtin_amdgcn_wmma_f32_16x16x32_f16(false, a, false, b,
                                                     (short)0, acc, false, false);
      }
      lstm_update(acc, c1, D1, hf, mBase, nBase, lane);
    }
    __syncthreads();
  }

  // ---- final FC on last-step h1 (kept in f32) ----
  if (tid < BT) {
    float s = bfc[0];
#pragma unroll 5
    for (int j = 0; j < H; ++j) s += h1f[tid * CS + j] * wfc[j];
    out[bBase + tid] = s;
  }
}

extern "C" void kernel_launch(void* const* d_in, const int* in_sizes, int n_in,
                              void* d_out, int out_size, void* d_ws, size_t ws_size,
                              hipStream_t stream) {
  (void)n_in; (void)d_ws; (void)ws_size;
  const float* x    = (const float*)d_in[0];
  const float* Wih0 = (const float*)d_in[1];
  const float* Whh0 = (const float*)d_in[2];
  const float* bih0 = (const float*)d_in[3];
  const float* bhh0 = (const float*)d_in[4];
  const float* Wih1 = (const float*)d_in[5];
  const float* Whh1 = (const float*)d_in[6];
  const float* bih1 = (const float*)d_in[7];
  const float* bhh1 = (const float*)d_in[8];
  const float* Wfc  = (const float*)d_in[9];
  const float* bfc  = (const float*)d_in[10];
  float* out = (float*)d_out;

  const int B = out_size;               // 2048
  const int T = in_sizes[0] / B;        // 512

  (void)hipFuncSetAttribute((const void*)lstm2_fused_kernel,
                            hipFuncAttributeMaxDynamicSharedMemorySize, SMEM_BYTES);

  dim3 grid(B / BT), block(NTHREADS);
  lstm2_fused_kernel<<<grid, block, SMEM_BYTES, stream>>>(
      x, Wih0, Whh0, bih0, bhh0, Wih1, Whh1, bih1, bhh1, Wfc, bfc, out, T);
}
